// EncodeProcessDecode_27754078666864
// MI455X (gfx1250) — matrix operand order repaired and checked by reading, not compiled, not run on hardware
//
#include <hip/hip_runtime.h>
#include <cstdint>
#include <cstddef>

// ---------------------------------------------------------------------------
// Types for CDNA5 (gfx1250) WMMA: wave32, v_wmma_f32_16x16x32_bf16
// ---------------------------------------------------------------------------
typedef __bf16 bf16;
typedef __attribute__((ext_vector_type(16))) __bf16 v16bf;
typedef __attribute__((ext_vector_type(8)))  float  v8f;
typedef __attribute__((ext_vector_type(4)))  unsigned int u32x4;
typedef __attribute__((ext_vector_type(2)))  unsigned int u32x2;
typedef __attribute__((ext_vector_type(4)))  float  f32x4;
typedef int v4i __attribute__((vector_size(16)));   // matches async builtin param type

union FragAB { v16bf bf; u32x4 u[2]; };
union Pack4  { bf16 h[4]; u32x2 u; };

#if __has_builtin(__builtin_amdgcn_global_load_async_to_lds_b128) && \
    __has_builtin(__builtin_amdgcn_s_wait_asynccnt)
#define USE_ASYNC_LDS 1
#define __AS1 __attribute__((address_space(1)))
#define __AS3 __attribute__((address_space(3)))
#else
#define USE_ASYNC_LDS 0
#endif

static __device__ __forceinline__ void async_or_copy_b128(const bf16* __restrict__ g,
                                                          bf16* __restrict__ l) {
#if USE_ASYNC_LDS
  __builtin_amdgcn_global_load_async_to_lds_b128(
      (__AS1 v4i*)(void*)const_cast<bf16*>(g),
      (__AS3 v4i*)(void*)l, 0, 0);
#else
  *(u32x4*)l = *(const u32x4*)g;
#endif
}

static __device__ __forceinline__ void async_fence() {
#if USE_ASYNC_LDS
  __builtin_amdgcn_s_wait_asynccnt(0);
#endif
}

static __device__ __forceinline__ u32x2 pack4_bf16(f32x4 f) {
  Pack4 p;
  p.h[0] = (bf16)f.x; p.h[1] = (bf16)f.y; p.h[2] = (bf16)f.z; p.h[3] = (bf16)f.w;
  return p.u;
}

// ---------------------------------------------------------------------------
// Wave-level GEMM with MT row-tiles: D[MT*16 x 128] (f32) =
//   A[MT*16 x 32*KT] (bf16 LDS, row stride ASTRIDE) x Wt (bf16 [128][KPAD]).
// Each B-fragment load feeds MT WMMAs (amortizes weight traffic).
//
// A-frag (16-bit A 16x32): lane {h=lane>>4, m=lane&15};
//   elems 0..7 = K(8h..8h+7), elems 8..15 = K(16+8h..23+8h)
// B-frag (16-bit B 32x16): lane {h, n}; same K order as A-frag, col n.
// D: acc[mt][nt][v] = row (mt*16 + v + 8h), col (nt*16 + n).
// ---------------------------------------------------------------------------
template<int KT, int ASTRIDE, int KPAD, int MT>
static __device__ __forceinline__ void wave_gemm_mt(const bf16* __restrict__ As,
                                                    const bf16* __restrict__ Wt,
                                                    v8f acc[][8])
{
  const int lane = threadIdx.x;
  const int h    = lane >> 4;
  const int mn   = lane & 15;
#pragma unroll
  for (int kt = 0; kt < KT; ++kt) {
    FragAB a[MT];
#pragma unroll
    for (int mt = 0; mt < MT; ++mt) {
      const bf16* ap = As + (mt * 16 + mn) * ASTRIDE + kt * 32;
      a[mt].u[0] = *(const u32x4*)(ap + 8 * h);
      a[mt].u[1] = *(const u32x4*)(ap + 16 + 8 * h);
    }
#pragma unroll
    for (int nt = 0; nt < 8; ++nt) {
      FragAB b;
      const bf16* bp = Wt + (size_t)(nt * 16 + mn) * KPAD + kt * 32;
      b.u[0] = *(const u32x4*)(bp + 8 * h);
      b.u[1] = *(const u32x4*)(bp + 16 + 8 * h);
#pragma unroll
      for (int mt = 0; mt < MT; ++mt)
        acc[mt][nt] = __builtin_amdgcn_wmma_f32_16x16x32_bf16(
            false, a[mt].bf, false, b.bf, (short)0, acc[mt][nt], false, false);
    }
  }
}

static __device__ __forceinline__ void bias_relu(v8f* acc, const float* __restrict__ b) {
  const int n = threadIdx.x & 15;
#pragma unroll
  for (int nt = 0; nt < 8; ++nt) {
    const float bb = b[nt * 16 + n];
#pragma unroll
    for (int v = 0; v < 8; ++v) {
      float x = acc[nt][v] + bb;
      acc[nt][v] = x > 0.f ? x : 0.f;
    }
  }
}

// Store one 16-row tile of acc to LDS H tile (row stride 128), as bf16.
static __device__ __forceinline__ void acc_to_lds_bf16(const v8f* acc, bf16* __restrict__ H) {
  const int lane = threadIdx.x, h = lane >> 4, n = lane & 15;
#pragma unroll
  for (int nt = 0; nt < 8; ++nt)
#pragma unroll
    for (int v = 0; v < 8; ++v)
      H[(v + 8 * h) * 128 + nt * 16 + n] = (bf16)acc[nt][v];
}

// Row-wise LayerNorm over 128 cols; rows live in 16-lane halves -> xor butterfly.
static __device__ __forceinline__ void layernorm_acc(v8f* acc,
                                                     const float* __restrict__ g,
                                                     const float* __restrict__ be) {
  const int n = threadIdx.x & 15;
  float gv[8], bv[8];
#pragma unroll
  for (int nt = 0; nt < 8; ++nt) { gv[nt] = g[nt * 16 + n]; bv[nt] = be[nt * 16 + n]; }
#pragma unroll
  for (int v = 0; v < 8; ++v) {
    float s = 0.f, s2 = 0.f;
#pragma unroll
    for (int nt = 0; nt < 8; ++nt) { float x = acc[nt][v]; s += x; s2 += x * x; }
#pragma unroll
    for (int off = 1; off < 16; off <<= 1) {
      s  += __shfl_xor(s,  off, 32);
      s2 += __shfl_xor(s2, off, 32);
    }
    const float mu   = s * 0.0078125f;           // /128
    const float var  = s2 * 0.0078125f - mu * mu;
    const float rinv = rsqrtf(var + 1e-5f);
#pragma unroll
    for (int nt = 0; nt < 8; ++nt)
      acc[nt][v] = (acc[nt][v] - mu) * rinv * gv[nt] + bv[nt];
  }
}

// ---------------------------------------------------------------------------
// Kernels
// ---------------------------------------------------------------------------
__global__ __launch_bounds__(256) void zero_f32_kernel(float* __restrict__ p, int n) {
  int i = blockIdx.x * 256 + threadIdx.x;
  if (i < n) p[i] = 0.f;
}

// Transpose + bf16 + zero-pad weights: src [K][Nout] f32 -> dst [Nout][Kpad] bf16
__global__ __launch_bounds__(256) void conv_wt_kernel(const float* __restrict__ src,
                                                      bf16* __restrict__ dst,
                                                      int K, int Kpad, int Nout) {
  int idx = blockIdx.x * 256 + threadIdx.x;
  if (idx >= Nout * Kpad) return;
  int n = idx / Kpad, k = idx - n * Kpad;
  dst[idx] = (bf16)((k < K) ? src[k * Nout + n] : 0.f);
}

// Node encoder: raw feats (6, pad 32) -> MLP -> LN -> nl (f32 + bf16)
__global__ __launch_bounds__(32) void node_encode_kernel(
    const int* __restrict__ node_type, const float* __restrict__ pvf,
    const float* __restrict__ matD, const float* __restrict__ matX,
    const float* __restrict__ nmean, const float* __restrict__ nstd,
    const bf16* __restrict__ w1t, const float* __restrict__ b1,
    const bf16* __restrict__ w2t, const float* __restrict__ b2,
    const float* __restrict__ g, const float* __restrict__ be,
    float* __restrict__ nl, bf16* __restrict__ nlb)
{
  __shared__ bf16 Xs[16 * 32];
  __shared__ bf16 Hs[16 * 128];
  const int lane = threadIdx.x;
  const int i0 = blockIdx.x * 16;
  if (lane < 16) {
    const int i = i0 + lane;
    float f[6];
    f[0] = pvf[i]; f[1] = matD[i]; f[2] = matX[i];
    const int t = node_type[i];
    f[3] = (t == 0) ? 1.f : 0.f; f[4] = (t == 1) ? 1.f : 0.f; f[5] = (t == 2) ? 1.f : 0.f;
#pragma unroll
    for (int k = 0; k < 6; ++k) Xs[lane * 32 + k] = (bf16)((f[k] - nmean[k]) / nstd[k]);
#pragma unroll
    for (int k = 6; k < 32; ++k) Xs[lane * 32 + k] = (bf16)0.f;
  }
  __syncthreads();
  v8f acc[1][8] = {};
  wave_gemm_mt<1, 32, 32, 1>(Xs, w1t, acc);
  bias_relu(acc[0], b1);
  acc_to_lds_bf16(acc[0], Hs);
  __syncthreads();
  v8f acc2[1][8] = {};
  wave_gemm_mt<4, 128, 128, 1>(Hs, w2t, acc2);
  bias_relu(acc2[0], b2);
  layernorm_acc(acc2[0], g, be);
  const int h = lane >> 4, n = lane & 15;
#pragma unroll
  for (int v = 0; v < 8; ++v) {
    const int i = i0 + v + 8 * h;
#pragma unroll
    for (int nt = 0; nt < 8; ++nt) {
      const int col = nt * 16 + n;
      const float x = acc2[0][nt][v];
      nl[(size_t)i * 128 + col]  = x;
      nlb[(size_t)i * 128 + col] = (bf16)x;
    }
  }
}

// Edge encoder: raw feats (7, pad 32) -> MLP -> LN -> el (f32)
__global__ __launch_bounds__(32) void edge_encode_kernel(
    const int* __restrict__ senders, const int* __restrict__ receivers,
    const float* __restrict__ mesh_pos, const float* __restrict__ world_pos,
    const float* __restrict__ pvf,
    const float* __restrict__ emean, const float* __restrict__ estd,
    const bf16* __restrict__ w1t, const float* __restrict__ b1,
    const bf16* __restrict__ w2t, const float* __restrict__ b2,
    const float* __restrict__ g, const float* __restrict__ be,
    float* __restrict__ el)
{
  __shared__ bf16 Xs[16 * 32];
  __shared__ bf16 Hs[16 * 128];
  const int lane = threadIdx.x;
  const int e0 = blockIdx.x * 16;
  if (lane < 16) {
    const int e = e0 + lane;
    const int s = senders[e], r = receivers[e];
    const float rmx = mesh_pos[2 * s]     - mesh_pos[2 * r];
    const float rmy = mesh_pos[2 * s + 1] - mesh_pos[2 * r + 1];
    const float nrm = sqrtf(rmx * rmx + rmy * rmy);
    const float rwx = world_pos[2 * s]     - world_pos[2 * r];
    const float rwy = world_pos[2 * s + 1] - world_pos[2 * r + 1];
    const float dp  = pvf[s] - pvf[r];
    float f[7] = {rmx, rmy, nrm, rwx, rwy, nrm, dp};
#pragma unroll
    for (int k = 0; k < 7; ++k) Xs[lane * 32 + k] = (bf16)((f[k] - emean[k]) / estd[k]);
#pragma unroll
    for (int k = 7; k < 32; ++k) Xs[lane * 32 + k] = (bf16)0.f;
  }
  __syncthreads();
  v8f acc[1][8] = {};
  wave_gemm_mt<1, 32, 32, 1>(Xs, w1t, acc);
  bias_relu(acc[0], b1);
  acc_to_lds_bf16(acc[0], Hs);
  __syncthreads();
  v8f acc2[1][8] = {};
  wave_gemm_mt<4, 128, 128, 1>(Hs, w2t, acc2);
  bias_relu(acc2[0], b2);
  layernorm_acc(acc2[0], g, be);
  const int h = lane >> 4, n = lane & 15;
#pragma unroll
  for (int v = 0; v < 8; ++v) {
    const int e = e0 + v + 8 * h;
#pragma unroll
    for (int nt = 0; nt < 8; ++nt)
      el[(size_t)e * 128 + nt * 16 + n] = acc2[0][nt][v];
  }
}

// GraphNet edge update, 32 edges per wave (MT=2):
// X = [nl[s] | nl[r] | el] (32 x 384 bf16); MLP(384->128->128)+LN;
// el += y; atomicAdd aggr[recv] += y.
__global__ __launch_bounds__(32) void edge_mlp_kernel(
    const int* __restrict__ senders, const int* __restrict__ receivers,
    const bf16* __restrict__ nlb, float* __restrict__ el, float* __restrict__ aggr,
    const bf16* __restrict__ w1t, const float* __restrict__ b1,
    const bf16* __restrict__ w2t, const float* __restrict__ b2,
    const float* __restrict__ g, const float* __restrict__ be)
{
  __shared__ bf16 Xs[32 * 384];
  __shared__ bf16 Hs[32 * 128];
  const int lane = threadIdx.x;
  const int e0 = blockIdx.x * 32;
  // gather nl[senders], nl[receivers] as bf16 -> LDS (async path if available)
  for (int t = lane; t < 512; t += 32) {
    const int row = t >> 4, c = (t & 15) * 8;
    const int s = senders[e0 + row];
    const int r = receivers[e0 + row];
    async_or_copy_b128(nlb + (size_t)s * 128 + c, Xs + row * 384 + c);
    async_or_copy_b128(nlb + (size_t)r * 128 + c, Xs + row * 384 + 128 + c);
  }
  // stream el rows (contiguous), convert f32 -> bf16 (overlaps with async gathers)
  for (int t = lane; t < 1024; t += 32) {
    const int row = t >> 5, c = (t & 31) * 4;
    f32x4 f = *(const f32x4*)(el + (size_t)(e0 + row) * 128 + c);
    *(u32x2*)(Xs + row * 384 + 256 + c) = pack4_bf16(f);
  }
  __builtin_prefetch(el + (size_t)(e0 + 32) * 128, 0, 3);
  async_fence();
  __syncthreads();
  v8f acc[2][8] = {};
  wave_gemm_mt<12, 384, 384, 2>(Xs, w1t, acc);
#pragma unroll
  for (int mt = 0; mt < 2; ++mt) {
    bias_relu(acc[mt], b1);
    acc_to_lds_bf16(acc[mt], Hs + mt * 16 * 128);
  }
  __syncthreads();
  v8f acc2[2][8] = {};
  wave_gemm_mt<4, 128, 128, 2>(Hs, w2t, acc2);
  const int h = lane >> 4, n = lane & 15;
#pragma unroll
  for (int mt = 0; mt < 2; ++mt) {
    bias_relu(acc2[mt], b2);
    layernorm_acc(acc2[mt], g, be);
#pragma unroll
    for (int v = 0; v < 8; ++v) {
      const int e = e0 + mt * 16 + v + 8 * h;
      const int r = receivers[e];
      float* erow = el   + (size_t)e * 128;
      float* arow = aggr + (size_t)r * 128;
#pragma unroll
      for (int nt = 0; nt < 8; ++nt) {
        const int col = nt * 16 + n;
        const float y = acc2[mt][nt][v];
        erow[col] += y;                 // residual (exclusive ownership)
        atomicAdd(arow + col, y);       // segment_sum scatter
      }
    }
  }
}

// GraphNet node update, 32 nodes per wave (MT=2):
// X = [nl | aggr] (32 x 256 bf16); MLP(256->128->128)+LN; nl += y; refresh nlb.
__global__ __launch_bounds__(32) void node_mlp_kernel(
    float* __restrict__ nl, bf16* __restrict__ nlb, const float* __restrict__ aggr,
    const bf16* __restrict__ w1t, const float* __restrict__ b1,
    const bf16* __restrict__ w2t, const float* __restrict__ b2,
    const float* __restrict__ g, const float* __restrict__ be)
{
  __shared__ bf16 Xs[32 * 256];
  __shared__ bf16 Hs[32 * 128];
  const int lane = threadIdx.x;
  const int i0 = blockIdx.x * 32;
  for (int t = lane; t < 512; t += 32) {
    const int row = t >> 4, c = (t & 15) * 8;
    async_or_copy_b128(nlb + (size_t)(i0 + row) * 128 + c, Xs + row * 256 + c);
  }
  for (int t = lane; t < 1024; t += 32) {
    const int row = t >> 5, c = (t & 31) * 4;
    f32x4 f = *(const f32x4*)(aggr + (size_t)(i0 + row) * 128 + c);
    *(u32x2*)(Xs + row * 256 + 128 + c) = pack4_bf16(f);
  }
  async_fence();
  __syncthreads();
  v8f acc[2][8] = {};
  wave_gemm_mt<8, 256, 256, 2>(Xs, w1t, acc);
#pragma unroll
  for (int mt = 0; mt < 2; ++mt) {
    bias_relu(acc[mt], b1);
    acc_to_lds_bf16(acc[mt], Hs + mt * 16 * 128);
  }
  __syncthreads();
  v8f acc2[2][8] = {};
  wave_gemm_mt<4, 128, 128, 2>(Hs, w2t, acc2);
  const int h = lane >> 4, n = lane & 15;
#pragma unroll
  for (int mt = 0; mt < 2; ++mt) {
    bias_relu(acc2[mt], b2);
    layernorm_acc(acc2[mt], g, be);
#pragma unroll
    for (int v = 0; v < 8; ++v) {
      const int i = i0 + mt * 16 + v + 8 * h;
      float* row = nl  + (size_t)i * 128;
      bf16* rowb = nlb + (size_t)i * 128;
#pragma unroll
      for (int nt = 0; nt < 8; ++nt) {
        const int col = nt * 16 + n;
        const float x = row[col] + acc2[mt][nt][v];
        row[col]  = x;
        rowb[col] = (bf16)x;
      }
    }
  }
}

// Decoder: nl[128] -> 8 (swish) -> 15; out[tw][i][os] = dec[tw*3+os] * (tw+1)
__global__ __launch_bounds__(256) void decode_kernel(
    const float* __restrict__ nl,
    const float* __restrict__ w1, const float* __restrict__ b1,
    const float* __restrict__ w2, const float* __restrict__ b2,
    float* __restrict__ out, int N)
{
  const int i = blockIdx.x * 256 + threadIdx.x;
  if (i >= N) return;
  const float* row = nl + (size_t)i * 128;
  float hbuf[8];
#pragma unroll
  for (int j = 0; j < 8; ++j) hbuf[j] = b1[j];
  for (int k = 0; k < 128; k += 4) {
    f32x4 x = *(const f32x4*)(row + k);
#pragma unroll
    for (int j = 0; j < 8; ++j)
      hbuf[j] += x.x * w1[k * 8 + j] + x.y * w1[(k + 1) * 8 + j] +
                 x.z * w1[(k + 2) * 8 + j] + x.w * w1[(k + 3) * 8 + j];
  }
#pragma unroll
  for (int j = 0; j < 8; ++j) hbuf[j] = hbuf[j] / (1.f + __expf(-hbuf[j])); // swish
#pragma unroll
  for (int t = 0; t < 15; ++t) {
    float d = b2[t];
#pragma unroll
    for (int j = 0; j < 8; ++j) d += hbuf[j] * w2[j * 15 + t];
    const int tw = t / 3, os = t - tw * 3;
    out[(size_t)tw * N * 3 + (size_t)i * 3 + os] = d * (float)(tw + 1);
  }
}

// ---------------------------------------------------------------------------
// Host driver
// ---------------------------------------------------------------------------
extern "C" void kernel_launch(void* const* d_in, const int* in_sizes, int n_in,
                              void* d_out, int out_size, void* d_ws, size_t ws_size,
                              hipStream_t stream) {
  (void)in_sizes; (void)n_in; (void)out_size; (void)ws_size;
  const int N = 20000, E = 120000, L = 128, MPS = 15;
  const int NT16 = N / 16, ET16 = E / 16;
  const int NT32 = N / 32, ET32 = E / 32;

  const int*   senders   = (const int*)d_in[0];
  const int*   receivers = (const int*)d_in[1];
  const int*   node_type = (const int*)d_in[2];
  const float* pvf       = (const float*)d_in[3];
  const float* matD      = (const float*)d_in[4];
  const float* matX      = (const float*)d_in[5];
  const float* mesh_pos  = (const float*)d_in[6];
  const float* world_pos = (const float*)d_in[7];
  const float* node_mean = (const float*)d_in[8];
  const float* node_std  = (const float*)d_in[9];
  const float* edge_mean = (const float*)d_in[10];
  const float* edge_std  = (const float*)d_in[11];
  const float* ne_w1 = (const float*)d_in[12]; const float* ne_b1 = (const float*)d_in[13];
  const float* ne_w2 = (const float*)d_in[14]; const float* ne_b2 = (const float*)d_in[15];
  const float* ne_g  = (const float*)d_in[16]; const float* ne_be = (const float*)d_in[17];
  const float* ee_w1 = (const float*)d_in[18]; const float* ee_b1 = (const float*)d_in[19];
  const float* ee_w2 = (const float*)d_in[20]; const float* ee_b2 = (const float*)d_in[21];
  const float* ee_g  = (const float*)d_in[22]; const float* ee_be = (const float*)d_in[23];
  const float* be_w1 = (const float*)d_in[24]; const float* be_b1 = (const float*)d_in[25];
  const float* be_w2 = (const float*)d_in[26]; const float* be_b2 = (const float*)d_in[27];
  const float* be_g  = (const float*)d_in[28]; const float* be_be = (const float*)d_in[29];
  const float* bn_w1 = (const float*)d_in[30]; const float* bn_b1 = (const float*)d_in[31];
  const float* bn_w2 = (const float*)d_in[32]; const float* bn_b2 = (const float*)d_in[33];
  const float* bn_g  = (const float*)d_in[34]; const float* bn_be = (const float*)d_in[35];
  const float* d_w1  = (const float*)d_in[36]; const float* d_b1  = (const float*)d_in[37];
  const float* d_w2  = (const float*)d_in[38]; const float* d_b2  = (const float*)d_in[39];

  // workspace carve-out
  char* wsp = (char*)d_ws;
  auto carve = [&](size_t bytes) -> void* {
    void* p = (void*)wsp;
    wsp += (bytes + 255) & ~(size_t)255;
    return p;
  };
  float* nl    = (float*)carve((size_t)N * L * 4);
  bf16*  nlb   = (bf16*) carve((size_t)N * L * 2);
  float* el    = (float*)carve((size_t)E * L * 4);
  float* aggr  = (float*)carve((size_t)N * L * 4);
  bf16* ne_w1t = (bf16*)carve(128 * 32 * 2);
  bf16* ne_w2t = (bf16*)carve(128 * 128 * 2);
  bf16* ee_w1t = (bf16*)carve(128 * 32 * 2);
  bf16* ee_w2t = (bf16*)carve(128 * 128 * 2);
  bf16* be_w1t = (bf16*)carve((size_t)MPS * 128 * 384 * 2);
  bf16* be_w2t = (bf16*)carve((size_t)MPS * 128 * 128 * 2);
  bf16* bn_w1t = (bf16*)carve((size_t)MPS * 128 * 256 * 2);
  bf16* bn_w2t = (bf16*)carve((size_t)MPS * 128 * 128 * 2);

  auto convW = [&](const float* src, bf16* dst, int K, int Kpad) {
    const int total = 128 * Kpad;
    conv_wt_kernel<<<(total + 255) / 256, 256, 0, stream>>>(src, dst, K, Kpad, 128);
  };
  convW(ne_w1, ne_w1t, 6, 32);
  convW(ne_w2, ne_w2t, 128, 128);
  convW(ee_w1, ee_w1t, 7, 32);
  convW(ee_w2, ee_w2t, 128, 128);
  for (int s = 0; s < MPS; ++s) {
    convW(be_w1 + (size_t)s * 384 * 128, be_w1t + (size_t)s * 128 * 384, 384, 384);
    convW(be_w2 + (size_t)s * 128 * 128, be_w2t + (size_t)s * 128 * 128, 128, 128);
    convW(bn_w1 + (size_t)s * 256 * 128, bn_w1t + (size_t)s * 128 * 256, 256, 256);
    convW(bn_w2 + (size_t)s * 128 * 128, bn_w2t + (size_t)s * 128 * 128, 128, 128);
  }

  // encode
  node_encode_kernel<<<NT16, 32, 0, stream>>>(node_type, pvf, matD, matX,
      node_mean, node_std, ne_w1t, ne_b1, ne_w2t, ne_b2, ne_g, ne_be, nl, nlb);
  edge_encode_kernel<<<ET16, 32, 0, stream>>>(senders, receivers, mesh_pos, world_pos,
      pvf, edge_mean, edge_std, ee_w1t, ee_b1, ee_w2t, ee_b2, ee_g, ee_be, el);

  // process: 15 GraphNet blocks
  const int AGGR_ELEMS = N * L;
  for (int s = 0; s < MPS; ++s) {
    zero_f32_kernel<<<(AGGR_ELEMS + 255) / 256, 256, 0, stream>>>(aggr, AGGR_ELEMS);
    edge_mlp_kernel<<<ET32, 32, 0, stream>>>(senders, receivers, nlb, el, aggr,
        be_w1t + (size_t)s * 128 * 384, be_b1 + (size_t)s * 128,
        be_w2t + (size_t)s * 128 * 128, be_b2 + (size_t)s * 128,
        be_g + (size_t)s * 128, be_be + (size_t)s * 128);
    node_mlp_kernel<<<NT32, 32, 0, stream>>>(nl, nlb, aggr,
        bn_w1t + (size_t)s * 128 * 256, bn_b1 + (size_t)s * 128,
        bn_w2t + (size_t)s * 128 * 128, bn_b2 + (size_t)s * 128,
        bn_g + (size_t)s * 128, bn_be + (size_t)s * 128);
  }

  // decode
  decode_kernel<<<(N + 255) / 256, 256, 0, stream>>>(nl, d_w1, d_b1, d_w2, d_b2,
                                                     (float*)d_out, N);
}
